// TwoStreamBlock_13280038879996
// MI455X (gfx1250) — matrix operand order repaired and checked by reading, not compiled, not run on hardware
//
#include <hip/hip_runtime.h>
#include <hip/hip_bf16.h>
#include <stdint.h>

// ---------------------------------------------------------------------------
// Types / helpers
// ---------------------------------------------------------------------------
typedef __attribute__((ext_vector_type(16))) __bf16 bf16x16;
typedef __attribute__((ext_vector_type(8)))  float  f32x8;
typedef __attribute__((ext_vector_type(4)))  int    i32x4;
typedef __attribute__((ext_vector_type(8)))  int    i32x8;

union BfFrag { bf16x16 v; unsigned u[8]; };

__device__ __forceinline__ unsigned short f32_to_bf16(float f) {
  union { float f; unsigned u; } v; v.f = f;
  unsigned r = v.u + 0x7FFFu + ((v.u >> 16) & 1u);   // round-to-nearest-even
  return (unsigned short)(r >> 16);
}

__device__ __forceinline__ f32x8 wmma_bf16(bf16x16 a, bf16x16 b, f32x8 c) {
  // D = A(16x32 bf16) * B(32x16 bf16) + C(16x16 f32)
  return __builtin_amdgcn_wmma_f32_16x16x32_bf16(
      /*neg_a=*/false, a, /*neg_b=*/false, b,
      /*c_mod=*/(short)0, c, /*reuse_a=*/false, /*reuse_b=*/false);
}

// A-fragment K index for dword i (ISA 7.12.2, 16-bit A 16x32 layout)
__device__ __forceinline__ int a_k0(int i, int half) {
  return ((i < 4) ? 2 * i : 16 + 2 * (i - 4)) + 8 * half;
}

// Low 32 bits of a generic LDS pointer = workgroup-relative LDS byte address.
__device__ __forceinline__ unsigned lds_off(const void* p) {
  return (unsigned)(unsigned long long)p;
}

// ---------------------------------------------------------------------------
// Tensor Data Mover: 2D bf16 tile, global -> LDS, with LDS row padding.
//   tile_w  : elements per row, tile_h : rows, row_stride : elements.
//   pad codes per D# group1 (interval: 0=2..7=256 DWORDs; amount: n -> n+1 DW).
// D# built per CDNA5 ISA; 2-group form (2D tensor).
// ---------------------------------------------------------------------------
__device__ __forceinline__ void tdm_load_tile2d_bf16(
    unsigned lds_byte, const unsigned short* gsrc,
    unsigned tile_w, unsigned tile_h, unsigned row_stride_elems,
    unsigned pad_interval_code, unsigned pad_amount_code) {
  unsigned long long ga = (unsigned long long)(const void*)gsrc;
  i32x4 g0;
  g0[0] = 1;                                             // count=1, user mode
  g0[1] = (int)lds_byte;                                 // lds_addr
  g0[2] = (int)(unsigned)ga;                             // global_addr[31:0]
  g0[3] = (int)(((unsigned)(ga >> 32) & 0x01FFFFFFu)     // global_addr[56:32]
                | (2u << 30));                           // type=2 ("image")
  i32x8 g1;
  g1[0] = (int)((1u << 16)                               // data_size=1 (2B)
                | (1u << 20)                             // pad_enable
                | (pad_interval_code << 22)
                | (pad_amount_code << 25));
  g1[1] = (int)((tile_w & 0xFFFFu) << 16);               // tensor_dim0[15:0]
  g1[2] = (int)(((tile_w >> 16) & 0xFFFFu)               // tensor_dim0[31:16]
                | ((tile_h & 0xFFFFu) << 16));           // tensor_dim1[15:0]
  g1[3] = (int)(((tile_h >> 16) & 0xFFFFu)               // tensor_dim1[31:16]
                | ((tile_w & 0xFFFFu) << 16));           // tile_dim0
  g1[4] = (int)(tile_h & 0xFFFFu);                       // tile_dim1 (dim2=0)
  g1[5] = (int)row_stride_elems;                         // dim0_stride[31:0]
  g1[6] = 0;                                             // stride hi / dim1_stride
  g1[7] = 0;
  asm volatile("tensor_load_to_lds %0, %1" :: "s"(g0), "s"(g1) : "memory");
}

#define DIMC      2048
#define N_HEADS   16
#define N_KVH     4
#define HEAD_DIM  128
#define ATTN_SCALE 0.12f
#define RMS_EPS   1.1920929e-07f

// ---------------------------------------------------------------------------
// Elementwise converts
// ---------------------------------------------------------------------------
__global__ void cvt_f32_bf16(const float* __restrict__ in,
                             unsigned short* __restrict__ out, long long n) {
  long long i = (long long)blockIdx.x * blockDim.x + threadIdx.x;
  long long stride = (long long)gridDim.x * blockDim.x;
  for (; i < n; i += stride) out[i] = f32_to_bf16(in[i]);
}

// kv_src = concat([xt, x0], seq axis) -> bf16 [B, 2T, D]
__global__ void build_kv_bf16(const float* __restrict__ xt,
                              const float* __restrict__ x0,
                              unsigned short* __restrict__ kv,
                              int B, int T, int D) {
  long long n = (long long)B * 2 * T * D;
  long long stride = (long long)gridDim.x * blockDim.x;
  for (long long i = (long long)blockIdx.x * blockDim.x + threadIdx.x; i < n; i += stride) {
    int col = (int)(i % D);
    long long row = i / D;
    int b = (int)(row / (2 * T));
    int s = (int)(row % (2 * T));
    float v = (s < T) ? xt[((long long)b * T + s) * D + col]
                      : x0[((long long)b * T + (s - T)) * D + col];
    kv[i] = f32_to_bf16(v);
  }
}

// ---------------------------------------------------------------------------
// GEMM: C[M,N] (f32) = A[M,K] (bf16, row-major) * W[N,K]^T (bf16, row-major)
// Block 256 threads = 8 waves; tile 128x128, K-step 32.
// TDM double-buffered operand staging: wave0 streams A tiles, wave1 W tiles;
// s_wait_tensorcnt(1) overlaps tile k+1 DMA with tile k WMMAs.
// ---------------------------------------------------------------------------
#define GA_STRIDE 40   // 32 + 8 pad bf16 elems (80B rows; TDM pad 16B / 64B)

__global__ __launch_bounds__(256) void gemm_bf16_nt(
    const unsigned short* __restrict__ A,
    const unsigned short* __restrict__ W,
    float* __restrict__ C, int M, int N, int K) {
  __shared__ unsigned short As[2][128 * GA_STRIDE];
  __shared__ unsigned short Ws[2][128 * GA_STRIDE];

  const int tid  = threadIdx.x;
  const int lane = tid & 31;
  const int wave = tid >> 5;
  const int half = lane >> 4;
  const int l16  = lane & 15;
  const int wm   = wave >> 1;      // 0..3
  const int wn   = wave & 1;       // 0..1
  const int mBase = blockIdx.x * 128;
  const int nBase = blockIdx.y * 128;

  f32x8 acc[2][4];
  #pragma unroll
  for (int ti = 0; ti < 2; ++ti)
    #pragma unroll
    for (int tj = 0; tj < 4; ++tj)
      #pragma unroll
      for (int e = 0; e < 8; ++e) acc[ti][tj][e] = 0.0f;

  const int nIter = K >> 5;
  // prologue: DMA tile 0
  if (wave == 0)
    tdm_load_tile2d_bf16(lds_off(&As[0][0]), A + (size_t)mBase * K,
                         32, 128, (unsigned)K, 3, 3);
  if (wave == 1)
    tdm_load_tile2d_bf16(lds_off(&Ws[0][0]), W + (size_t)nBase * K,
                         32, 128, (unsigned)K, 3, 3);

  for (int it = 0; it < nIter; ++it) {
    const int cur = it & 1;
    if (it + 1 < nIter) {
      // prefetch next tile into the other buffer while this one computes
      if (wave == 0)
        tdm_load_tile2d_bf16(lds_off(&As[cur ^ 1][0]),
                             A + (size_t)mBase * K + (it + 1) * 32,
                             32, 128, (unsigned)K, 3, 3);
      if (wave == 1)
        tdm_load_tile2d_bf16(lds_off(&Ws[cur ^ 1][0]),
                             W + (size_t)nBase * K + (it + 1) * 32,
                             32, 128, (unsigned)K, 3, 3);
      __builtin_amdgcn_s_wait_tensorcnt(1);   // tile `it` done; `it+1` in flight
    } else {
      __builtin_amdgcn_s_wait_tensorcnt(0);
    }
    __syncthreads();                          // tile `it` visible to all waves

    BfFrag afr[2], bfr[4];
    #pragma unroll
    for (int ti = 0; ti < 2; ++ti) {
      int rb = wm * 32 + ti * 16 + l16;
      #pragma unroll
      for (int i = 0; i < 8; ++i)
        afr[ti].u[i] = *(const unsigned*)(&As[cur][rb * GA_STRIDE + a_k0(i, half)]);
    }
    #pragma unroll
    for (int tj = 0; tj < 4; ++tj) {
      int cb = wn * 64 + tj * 16 + l16;
      #pragma unroll
      for (int i = 0; i < 8; ++i)
        bfr[tj].u[i] = *(const unsigned*)(&Ws[cur][cb * GA_STRIDE + a_k0(i, half)]);
    }
    #pragma unroll
    for (int ti = 0; ti < 2; ++ti)
      #pragma unroll
      for (int tj = 0; tj < 4; ++tj)
        acc[ti][tj] = wmma_bf16(afr[ti].v, bfr[tj].v, acc[ti][tj]);
    __syncthreads();                          // buffer free before reuse
  }

  // C/D layout: VGPR r -> row (r + 8*half), col = lane%16
  #pragma unroll
  for (int ti = 0; ti < 2; ++ti)
    #pragma unroll
    for (int tj = 0; tj < 4; ++tj) {
      int rowb = mBase + wm * 32 + ti * 16 + half * 8;
      int col  = nBase + wn * 64 + tj * 16 + l16;
      #pragma unroll
      for (int r = 0; r < 8; ++r)
        C[(size_t)(rowb + r) * N + col] = acc[ti][tj][r];
    }
}

// ---------------------------------------------------------------------------
// Fused per-head RMSNorm + RoPE, f32 in -> bf16 out.  One wave per (row,head).
// rows = B*seq, pos = row % seq.  HEAD_DIM=128 -> 4 elems/lane.
// ---------------------------------------------------------------------------
__global__ __launch_bounds__(256) void rmsrope_bf16(
    const float* __restrict__ x, unsigned short* __restrict__ out,
    int rows, int H, int seq) {
  int gw   = blockIdx.x * 8 + (threadIdx.x >> 5);
  int lane = threadIdx.x & 31;
  if (gw >= rows * H) return;
  int row = gw / H, h = gw % H;
  float pos = (float)(row % seq);

  const float* px = x + ((size_t)row * H + h) * HEAD_DIM;
  float4 v = *(const float4*)(px + lane * 4);

  float ss = v.x * v.x + v.y * v.y + v.z * v.z + v.w * v.w;
  #pragma unroll
  for (int off = 1; off < 32; off <<= 1) ss += __shfl_xor(ss, off, 32);
  float r = rsqrtf(ss * (1.0f / 128.0f) + RMS_EPS);
  v.x *= r; v.y *= r; v.z *= r; v.w *= r;

  // rotate-half partner exchange (lane ^ 16 holds the other half)
  float vv[4] = { v.x, v.y, v.z, v.w };
  float pp[4];
  #pragma unroll
  for (int e = 0; e < 4; ++e) pp[e] = __shfl_xor(vv[e], 16, 32);

  unsigned short* po = out + ((size_t)row * H + h) * HEAD_DIM;
  #pragma unroll
  for (int e = 0; e < 4; ++e) {
    int j = (lane & 15) * 4 + e;            // 0..63 (half-dim index)
    float c = 1.0f, s = 0.0f;
    if (j < 32) {
      float freq = exp2f(-10.0f * (float)j / 31.0f);   // (1/1024)^(j/31)
      float th = pos * freq;
      c = __cosf(th); s = __sinf(th);
    }
    float y = (lane < 16) ? (vv[e] * c + pp[e] * s)      // y1 =  x1*c + x2*s
                          : (-pp[e] * s + vv[e] * c);    // y2 = -x1*s + x2*c
    po[lane * 4 + e] = f32_to_bf16(y);
  }
}

// ---------------------------------------------------------------------------
// Flash attention: q[B,T,16,128] x k/v[B,S,4,128] (bf16) -> y[B,T,2048] (bf16)
// Block = 4 waves; each wave owns 16 query rows.  K chunk staged by TDM
// (wave 0) while all threads transpose-stage the V chunk; online softmax;
// 16 WMMAs per 32-column chunk per wave.
// ---------------------------------------------------------------------------
#define KS_STRIDE 136   // 128 + 8 pad bf16 (272B rows; TDM pad 16B / 256B)
#define VT_STRIDE 34    // 32 + 2 pad bf16, keeps (d*34+s)*2 4B-aligned

__global__ __launch_bounds__(128) void attention_flash(
    const unsigned short* __restrict__ q,
    const unsigned short* __restrict__ k,
    const unsigned short* __restrict__ v,
    unsigned short* __restrict__ y, int T, int S) {
  __shared__ unsigned short Ks[32 * KS_STRIDE];   // K chunk, row = s, col = d
  __shared__ unsigned short Vt[128 * VT_STRIDE];  // V chunk transposed [d][s]
  __shared__ float Pws[4][16 * 32];               // per-wave prob scratch

  const int tid  = threadIdx.x;
  const int lane = tid & 31;
  const int wave = tid >> 5;
  const int half = lane >> 4;
  const int l16  = lane & 15;
  const int b    = blockIdx.z;
  const int h    = blockIdx.y;
  const int kvh  = h >> 2;                        // GROUP_SIZE = 4
  const int t0   = blockIdx.x * 64 + wave * 16;

  // Q fragments: 16x128 = 4 A-frags of 16x32
  BfFrag qf[4];
  const unsigned short* qp = q + (((size_t)b * T + t0 + l16) * N_HEADS + h) * HEAD_DIM;
  #pragma unroll
  for (int c = 0; c < 4; ++c)
    #pragma unroll
    for (int i = 0; i < 8; ++i)
      qf[c].u[i] = *(const unsigned*)(qp + c * 32 + a_k0(i, half));

  f32x8 acc[8];                                   // y tile 16x128 f32
  #pragma unroll
  for (int dt = 0; dt < 8; ++dt)
    #pragma unroll
    for (int e = 0; e < 8; ++e) acc[dt][e] = 0.0f;
  float M[8], L[8];
  #pragma unroll
  for (int r = 0; r < 8; ++r) { M[r] = -3.0e38f; L[r] = 0.0f; }

  for (int s0 = 0; s0 < S; s0 += 32) {
    __syncthreads();                              // previous chunk fully read
    // K chunk via Tensor Data Mover (rows s0..s0+31, stride N_KVH*128)
    if (wave == 0)
      tdm_load_tile2d_bf16(lds_off(&Ks[0]),
                           k + (((size_t)b * S + s0) * N_KVH + kvh) * HEAD_DIM,
                           128, 32, N_KVH * HEAD_DIM, 5, 3);
    // V chunk transpose staging (overlaps the TDM): batch loads, then stores
    uint4 vtmp[4];
    #pragma unroll
    for (int j = 0; j < 4; ++j) {
      int c = tid + j * 128;
      int srow = c >> 4, part = c & 15;
      vtmp[j] = *(const uint4*)(v + (((size_t)b * S + s0 + srow) * N_KVH + kvh) * HEAD_DIM + part * 8);
    }
    #pragma unroll
    for (int j = 0; j < 4; ++j) {
      int c = tid + j * 128;
      int srow = c >> 4, part = c & 15;
      const unsigned short* e8 = (const unsigned short*)&vtmp[j];
      #pragma unroll
      for (int e = 0; e < 8; ++e)
        Vt[(part * 8 + e) * VT_STRIDE + srow] = e8[e];
    }
    __builtin_amdgcn_s_wait_tensorcnt(0);
    __syncthreads();

    // scores: two 16x16 tiles over K=128
    f32x8 s0v, s1v;
    #pragma unroll
    for (int e = 0; e < 8; ++e) { s0v[e] = 0.0f; s1v[e] = 0.0f; }
    #pragma unroll
    for (int c = 0; c < 4; ++c) {
      BfFrag bk0, bk1;
      #pragma unroll
      for (int i = 0; i < 8; ++i) {
        int kc = c * 32 + a_k0(i, half);
        bk0.u[i] = *(const unsigned*)(&Ks[(l16)      * KS_STRIDE + kc]);
        bk1.u[i] = *(const unsigned*)(&Ks[(16 + l16) * KS_STRIDE + kc]);
      }
      s0v = wmma_bf16(qf[c].v, bk0.v, s0v);
      s1v = wmma_bf16(qf[c].v, bk1.v, s1v);
    }

    // online softmax update (row r+8*half, cols in lanes of same half)
    #pragma unroll
    for (int r = 0; r < 8; ++r) {
      float a0 = s0v[r] * ATTN_SCALE;
      float a1 = s1v[r] * ATTN_SCALE;
      float mx = fmaxf(a0, a1);
      #pragma unroll
      for (int off = 1; off < 16; off <<= 1) mx = fmaxf(mx, __shfl_xor(mx, off, 32));
      float newM = fmaxf(M[r], mx);
      float corr = __expf(M[r] - newM);
      M[r] = newM;
      L[r] *= corr;
      #pragma unroll
      for (int dt = 0; dt < 8; ++dt) acc[dt][r] *= corr;
      float p0 = __expf(a0 - newM);
      float p1 = __expf(a1 - newM);
      float rs = p0 + p1;
      #pragma unroll
      for (int off = 1; off < 16; off <<= 1) rs += __shfl_xor(rs, off, 32);
      L[r] += rs;
      Pws[wave][(r + 8 * half) * 32 + l16]      = p0;
      Pws[wave][(r + 8 * half) * 32 + 16 + l16] = p1;
    }

    // P (16x32) as A-frag, via LDS transpose + bf16 pack
    BfFrag pf;
    #pragma unroll
    for (int i = 0; i < 8; ++i) {
      int k0 = a_k0(i, half);
      float pa = Pws[wave][l16 * 32 + k0];
      float pb = Pws[wave][l16 * 32 + k0 + 1];
      pf.u[i] = (unsigned)f32_to_bf16(pa) | ((unsigned)f32_to_bf16(pb) << 16);
    }

    // acc += P x V  (8 d-tiles of 16x16)
    #pragma unroll
    for (int dt = 0; dt < 8; ++dt) {
      BfFrag vf;
      #pragma unroll
      for (int i = 0; i < 8; ++i)
        vf.u[i] = *(const unsigned*)(&Vt[(dt * 16 + l16) * VT_STRIDE + a_k0(i, half)]);
      acc[dt] = wmma_bf16(pf.v, vf.v, acc[dt]);
    }
  }

  // normalize + store bf16 y[b, t, h*128 + d]
  #pragma unroll
  for (int r = 0; r < 8; ++r) {
    float inv = 1.0f / L[r];
    size_t base = ((size_t)b * T + t0 + r + 8 * half) * (size_t)DIMC + h * HEAD_DIM + l16;
    #pragma unroll
    for (int dt = 0; dt < 8; ++dt)
      y[base + dt * 16] = f32_to_bf16(acc[dt][r] * inv);
  }
}

// ---------------------------------------------------------------------------
// Launch
// ---------------------------------------------------------------------------
extern "C" void kernel_launch(void* const* d_in, const int* in_sizes, int n_in,
                              void* d_out, int out_size, void* d_ws, size_t ws_size,
                              hipStream_t stream) {
  const float* xt = (const float*)d_in[0];
  const float* x0 = (const float*)d_in[1];
  const float* wq = (const float*)d_in[2];
  const float* wk = (const float*)d_in[3];
  const float* wv = (const float*)d_in[4];
  const float* wo = (const float*)d_in[5];
  float* out = (float*)d_out;

  const int B = 2, T = 2048, D = DIMC, S = 2 * T;
  const int MQ = B * T;        // 4096 query rows
  const int MK = B * S;        // 8192 kv rows

  char* ws = (char*)d_ws;
  size_t off = 0;
  auto carve = [&](size_t bytes) -> void* {
    void* p = ws + off;
    off += (bytes + 255) & ~(size_t)255;
    return p;
  };
  unsigned short* xt_bf = (unsigned short*)carve((size_t)MQ * D * 2);
  unsigned short* kv_bf = (unsigned short*)carve((size_t)MK * D * 2);
  unsigned short* wq_bf = (unsigned short*)carve((size_t)D * D * 2);
  unsigned short* wk_bf = (unsigned short*)carve((size_t)512 * D * 2);
  unsigned short* wv_bf = (unsigned short*)carve((size_t)512 * D * 2);
  unsigned short* wo_bf = (unsigned short*)carve((size_t)D * D * 2);
  float* q_f32 = (float*)carve((size_t)MQ * D * 4);
  float* k_f32 = (float*)carve((size_t)MK * 512 * 4);
  float* v_f32 = (float*)carve((size_t)MK * 512 * 4);
  unsigned short* q_bf = (unsigned short*)carve((size_t)MQ * D * 2);
  unsigned short* k_bf = (unsigned short*)carve((size_t)MK * 512 * 2);
  unsigned short* v_bf = (unsigned short*)carve((size_t)MK * 512 * 2);
  unsigned short* y_bf = (unsigned short*)q_f32;  // reuse: q_f32 dead by then

  // 1) bf16 conversions
  cvt_f32_bf16<<<2048, 256, 0, stream>>>(wq, wq_bf, (long long)D * D);
  cvt_f32_bf16<<<1024, 256, 0, stream>>>(wk, wk_bf, (long long)512 * D);
  cvt_f32_bf16<<<1024, 256, 0, stream>>>(wv, wv_bf, (long long)512 * D);
  cvt_f32_bf16<<<2048, 256, 0, stream>>>(wo, wo_bf, (long long)D * D);
  cvt_f32_bf16<<<2048, 256, 0, stream>>>(xt, xt_bf, (long long)MQ * D);
  build_kv_bf16<<<4096, 256, 0, stream>>>(xt, x0, kv_bf, B, T, D);

  // 2) QKV projections (TDM-staged WMMA GEMMs)
  gemm_bf16_nt<<<dim3(MQ / 128, D / 128), 256, 0, stream>>>(xt_bf, wq_bf, q_f32, MQ, D, D);
  gemm_bf16_nt<<<dim3(MK / 128, 512 / 128), 256, 0, stream>>>(kv_bf, wk_bf, k_f32, MK, 512, D);
  gemm_bf16_nt<<<dim3(MK / 128, 512 / 128), 256, 0, stream>>>(kv_bf, wv_bf, v_f32, MK, 512, D);

  // 3) RMSNorm + RoPE (q, k); v straight to bf16
  rmsrope_bf16<<<(MQ * N_HEADS + 7) / 8, 256, 0, stream>>>(q_f32, q_bf, MQ, N_HEADS, T);
  rmsrope_bf16<<<(MK * N_KVH + 7) / 8, 256, 0, stream>>>(k_f32, k_bf, MK, N_KVH, S);
  cvt_f32_bf16<<<2048, 256, 0, stream>>>(v_f32, v_bf, (long long)MK * 512);

  // 4) Flash attention (TDM K staging, WMMA scores + WMMA PV, online softmax)
  attention_flash<<<dim3(T / 64, N_HEADS, B), 128, 0, stream>>>(q_bf, k_bf, v_bf, y_bf, T, S);

  // 5) Output projection -> f32 d_out
  gemm_bf16_nt<<<dim3(MQ / 128, D / 128), 256, 0, stream>>>(y_bf, wo_bf, out, MQ, D, D);
}
